// TEMRecurrent_6751688590059
// MI455X (gfx1250) — matrix operand order repaired and checked by reading, not compiled, not run on hardware
//
#include <hip/hip_runtime.h>

// ---------------------------------------------------------------------------
// TEM recurrent model for MI455X (gfx1250, wave32, WMMA).
// B=16, L=1024, V=10000, D=512, DG=DX=256, LAM=0.5, EPS=1e-5.
//
// Strategy: precompute gi = e @ W_ih^T for all B*L rows in one large bf16
// WMMA GEMM (M-blocked 4x per wave to lift arithmetic intensity on the
// L2-resident weight stream); per timestep run the h-dependent small GEMMs,
// gate math, Hebbian memory update + LayerNorm, and the logits GEMM as
// stream-ordered kernels (the serial chain makes this latency bound).
// ---------------------------------------------------------------------------

typedef __attribute__((ext_vector_type(16))) __bf16 v16bf;
typedef __attribute__((ext_vector_type(8)))  __bf16 v8bf;
typedef __attribute__((ext_vector_type(8)))  float  v8f;

#define B_  16
#define L_  1024
#define V_  10000
#define D_  512
#define DG_ 256
#define DX_ 256

__device__ __forceinline__ __bf16 f2bf(float f) {
  unsigned u = __builtin_bit_cast(unsigned, f);
  u += 0x7FFFu + ((u >> 16) & 1u);           // round-to-nearest-even
  unsigned short h = (unsigned short)(u >> 16);
  return __builtin_bit_cast(__bf16, h);
}

// ---------------------------------------------------------------------------
// C(Mtiles*16, N) = A(Mtiles*16, K) @ W(N, K)^T + bias
//   A row-major contiguous bf16; W row-major (N,K) bf16 so the B-fragment is
//   16 contiguous K values per lane (A @ W^T needs no transpose anywhere).
//   Each wave computes MT stacked 16x16 C tiles sharing one B fragment
//   (4x B-operand reuse for the big hoisted GEMM); 8 waves per block.
//   All MT A-fragments are loaded from LDS *before* the WMMA chain so the
//   scheduler can overlap ds_load latency with WMMA issue (partial DScnt
//   waits instead of a full drain per tile).
//   K <= 512, K % 32 == 0; Ntiles guarded per-wave (wave-uniform branch).
// ---------------------------------------------------------------------------
template <int MT>
__global__ void gemm_bf16_wmma(const __bf16* __restrict__ A,
                               const __bf16* __restrict__ W,
                               const float*  __restrict__ bias,
                               float* __restrict__ C, long ldc,
                               int K, int Ntiles) {
  __shared__ __bf16 lA[MT * 16 * (512 + 8)];   // +8 elem row pad vs bank conflicts
  const int lid  = threadIdx.x & 31;
  const int wave = threadIdx.x >> 5;
  const int Kp   = K + 8;

  // ---- stage A panel (MT*16 x K) into LDS, 16B vectors -----------------
  const __bf16* Apanel = A + (size_t)blockIdx.y * (MT * 16) * K;
  for (int c = threadIdx.x * 8; c < MT * 16 * K; c += blockDim.x * 8) {
    int row = c / K, col = c - row * K;
    *reinterpret_cast<v8bf*>(&lA[row * Kp + col]) =
        *reinterpret_cast<const v8bf*>(Apanel + c);
  }
  __syncthreads();

  const int tile_n = blockIdx.x * 8 + wave;
  if (tile_n >= Ntiles) return;                // wave-uniform; EXEC stays full

  // Fragment k order this code uses (contiguous k = 16h + s, same for A and B):
  //   A (16x32 bf16): lanes 0-15 -> M=lane, K 0..15;
  //                   lanes 16-31 -> M=lane-16, K 16..31.
  //   B (32x16 bf16): lanes 0-15 -> N=lane, K 0..15; lanes 16-31 -> K 16..31.
  const int m  = lid & 15;
  const int kb = (lid < 16) ? 0 : 16;
  const int nb = (lid < 16) ? 0 : 16;
  const int n  = tile_n * 16 + (lid & 15);
  const __bf16* wrow = W + (size_t)n * K;

  v8f acc[MT] = {};
  for (int k0 = 0; k0 < K; k0 += 32) {
    v16bf bfrag = *reinterpret_cast<const v16bf*>(wrow + k0 + nb);
    __builtin_prefetch(wrow + k0 + 64, 0, 3);  // near-scope prefetch of next slice

    // Issue ALL A-fragment LDS loads first ...
    v16bf afrag[MT];
#pragma unroll
    for (int mt = 0; mt < MT; ++mt) {
      const __bf16* ap = &lA[(mt * 16 + m) * Kp + k0 + kb];
      v8bf a0 = *reinterpret_cast<const v8bf*>(ap);
      v8bf a1 = *reinterpret_cast<const v8bf*>(ap + 8);
#pragma unroll
      for (int i = 0; i < 8; ++i) { afrag[mt][i] = a0[i]; afrag[mt][8 + i] = a1[i]; }
    }
    // ... then run the WMMA chain (B fragment reused MT times).
#pragma unroll
    for (int mt = 0; mt < MT; ++mt)
      acc[mt] = __builtin_amdgcn_wmma_f32_16x16x32_bf16(
          false, afrag[mt], false, bfrag, (short)0, acc[mt], false, false);
  }

  // C tile: VGPR r, lanes 0-15 -> M=r, lanes 16-31 -> M=8+r, N=lid&15.
  const float bv = bias ? bias[n] : 0.0f;
#pragma unroll
  for (int mt = 0; mt < MT; ++mt) {
    const int mrow = ((int)blockIdx.y * MT + mt) * 16 + ((lid < 16) ? 0 : 8);
#pragma unroll
    for (int r = 0; r < 8; ++r)
      C[(size_t)(mrow + r) * ldc + n] = acc[mt][r] + bv;
  }
}

// ---------------------------------------------------------------------------
// Elementwise / support kernels
// ---------------------------------------------------------------------------
__global__ void cvt_f32_bf16(const float* __restrict__ src,
                             __bf16* __restrict__ dst, int n) {
  for (int i = blockIdx.x * blockDim.x + threadIdx.x; i < n;
       i += gridDim.x * blockDim.x)
    dst[i] = f2bf(src[i]);
}

__global__ void concat3(const float* __restrict__ a, const float* __restrict__ b,
                        const float* __restrict__ c, float* __restrict__ dst) {
  int i = blockIdx.x * blockDim.x + threadIdx.x;
  if (i < 256)       dst[i] = a[i];
  else if (i < 512)  dst[i] = b[i - 256];
  else if (i < 768)  dst[i] = c[i - 512];
}

__global__ void zero_f32(float* __restrict__ p, int n) {
  for (int i = blockIdx.x * blockDim.x + threadIdx.x; i < n;
       i += gridDim.x * blockDim.x)
    p[i] = 0.0f;
}

__global__ void gather_embed(const int* __restrict__ tok,
                             const float* __restrict__ emb,
                             __bf16* __restrict__ E) {
  const size_t total = (size_t)B_ * L_ * D_;
  for (size_t i = (size_t)blockIdx.x * blockDim.x + threadIdx.x; i < total;
       i += (size_t)gridDim.x * blockDim.x) {
    size_t r = i >> 9;               // D_ == 512
    int    d = (int)(i & 511);
    E[i] = f2bf(emb[(size_t)tok[r] * D_ + d]);
  }
}

// GRU gate math for one timestep: 16x512 elements.
__global__ void gru_gates(const float* __restrict__ gi_all,
                          const float* __restrict__ gh,
                          float* __restrict__ h, __bf16* __restrict__ h_bf,
                          int t) {
  int idx = blockIdx.x * blockDim.x + threadIdx.x;
  if (idx >= B_ * D_) return;
  int b = idx >> 9, d = idx & 511;
  const float* gi  = gi_all + ((size_t)(b * L_ + t)) * (3 * D_);
  const float* ghb = gh + (size_t)b * (3 * D_);
  float r  = 1.0f / (1.0f + __expf(-(gi[d]          + ghb[d])));
  float z  = 1.0f / (1.0f + __expf(-(gi[D_ + d]     + ghb[D_ + d])));
  float nn = tanhf(gi[2 * D_ + d] + r * ghb[2 * D_ + d]);
  float ho = h[idx];
  float hn = (1.0f - z) * nn + z * ho;
  h[idx]    = hn;
  h_bf[idx] = f2bf(hn);
}

// Per-batch Hebbian memory read + write + LayerNorm; one block per batch row.
// gxp rows hold [g(256) | x(256) | xp(256)].
__global__ void mem_update(const float* __restrict__ gxp,
                           float* __restrict__ M,
                           const float* __restrict__ ln_g,
                           const float* __restrict__ ln_b,
                           __bf16* __restrict__ xn_bf) {
  const int b = blockIdx.x, xx = threadIdx.x;   // 256 threads
  __shared__ float sG[DG_], s1[DX_], s2[DX_];
  const float* row = gxp + (size_t)b * 768;
  sG[xx]   = row[xx];
  float xv = row[256 + xx];
  float xp = row[512 + xx];
  __syncthreads();

  float* Mb = M + (size_t)b * DG_ * DX_;
  float retr = 0.0f;
#pragma unroll 4
  for (int gg = 0; gg < DG_; ++gg) {
    float gv = sG[gg];
    float mv = Mb[gg * DX_ + xx];     // read M_{t-1}
    retr += gv * mv;
    Mb[gg * DX_ + xx] = mv + gv * xv; // Hebbian write
  }
  float xh = 0.5f * xp + 0.5f * retr; // LAM = 0.5

  s1[xx] = xh; s2[xx] = xh * xh;
  __syncthreads();
  for (int s = 128; s > 0; s >>= 1) {
    if (xx < s) { s1[xx] += s1[xx + s]; s2[xx] += s2[xx + s]; }
    __syncthreads();
  }
  float mu  = s1[0] * (1.0f / DX_);
  float var = s2[0] * (1.0f / DX_) - mu * mu;
  float xn  = (xh - mu) * rsqrtf(var + 1e-5f) * ln_g[xx] + ln_b[xx];
  xn_bf[b * DX_ + xx] = f2bf(xn);
}

// ---------------------------------------------------------------------------
// Workspace layout (bytes, all 256B-aligned)
// ---------------------------------------------------------------------------
static const size_t OFF_EBF  = 0;                               // 16384*512 bf16
static const size_t OFF_WIH  = OFF_EBF  + (size_t)16384*512*2;  // 1536*512 bf16
static const size_t OFF_WHH  = OFF_WIH  + (size_t)1536*512*2;
static const size_t OFF_WCAT = OFF_WHH  + (size_t)1536*512*2;   // 768*512 bf16
static const size_t OFF_WO   = OFF_WCAT + (size_t)768*512*2;    // 10000*256 bf16
static const size_t OFF_BCAT = OFF_WO   + (size_t)10000*256*2;  // 768 f32
static const size_t OFF_GI   = OFF_BCAT + (size_t)768*4;        // 16384*1536 f32
static const size_t OFF_GH   = OFF_GI   + (size_t)16384*1536*4; // 16*1536 f32
static const size_t OFF_H    = OFF_GH   + (size_t)16*1536*4;    // 16*512 f32
static const size_t OFF_HBF  = OFF_H    + (size_t)16*512*4;     // 16*512 bf16
static const size_t OFF_GXP  = OFF_HBF  + (size_t)16*512*2;     // 16*768 f32
static const size_t OFF_M    = OFF_GXP  + (size_t)16*768*4;     // 16*256*256 f32
static const size_t OFF_XNBF = OFF_M    + (size_t)16*256*256*4; // 16*256 bf16

extern "C" void kernel_launch(void* const* d_in, const int* in_sizes, int n_in,
                              void* d_out, int out_size, void* d_ws, size_t ws_size,
                              hipStream_t stream) {
  (void)in_sizes; (void)n_in; (void)out_size; (void)ws_size;

  const int*   tokens = (const int*)  d_in[0];
  const float* emb    = (const float*)d_in[1];
  const float* W_ih   = (const float*)d_in[2];
  const float* W_hh   = (const float*)d_in[3];
  const float* b_ih   = (const float*)d_in[4];
  const float* b_hh   = (const float*)d_in[5];
  const float* Wg     = (const float*)d_in[6];
  const float* bg     = (const float*)d_in[7];
  const float* Wx     = (const float*)d_in[8];
  const float* bx     = (const float*)d_in[9];
  const float* Wp     = (const float*)d_in[10];
  const float* bp     = (const float*)d_in[11];
  const float* ln_g   = (const float*)d_in[12];
  const float* ln_b   = (const float*)d_in[13];
  const float* Wo     = (const float*)d_in[14];
  const float* bo     = (const float*)d_in[15];
  float* out = (float*)d_out;

  char* ws = (char*)d_ws;
  __bf16* Ebf    = (__bf16*)(ws + OFF_EBF);
  __bf16* WihBf  = (__bf16*)(ws + OFF_WIH);
  __bf16* WhhBf  = (__bf16*)(ws + OFF_WHH);
  __bf16* WcatBf = (__bf16*)(ws + OFF_WCAT);
  __bf16* WoBf   = (__bf16*)(ws + OFF_WO);
  float*  bcat   = (float*) (ws + OFF_BCAT);
  float*  gi_all = (float*) (ws + OFF_GI);
  float*  gh     = (float*) (ws + OFF_GH);
  float*  h      = (float*) (ws + OFF_H);
  __bf16* h_bf   = (__bf16*)(ws + OFF_HBF);
  float*  gxp    = (float*) (ws + OFF_GXP);
  float*  Mbuf   = (float*) (ws + OFF_M);
  __bf16* xn_bf  = (__bf16*)(ws + OFF_XNBF);

  // ---- one-time (per launch) weight conversion & state init -------------
  cvt_f32_bf16<<<1024, 256, 0, stream>>>(W_ih, WihBf, 3 * D_ * D_);
  cvt_f32_bf16<<<1024, 256, 0, stream>>>(W_hh, WhhBf, 3 * D_ * D_);
  cvt_f32_bf16<<<512,  256, 0, stream>>>(Wg, WcatBf,               DG_ * D_);
  cvt_f32_bf16<<<512,  256, 0, stream>>>(Wx, WcatBf +     DG_*D_,  DX_ * D_);
  cvt_f32_bf16<<<512,  256, 0, stream>>>(Wp, WcatBf + 2 * DG_*D_,  DX_ * D_);
  cvt_f32_bf16<<<2048, 256, 0, stream>>>(Wo, WoBf, V_ * DX_);
  concat3<<<3, 256, 0, stream>>>(bg, bx, bp, bcat);
  zero_f32<<<2048, 256, 0, stream>>>(Mbuf, B_ * DG_ * DX_);
  zero_f32<<<32,   256, 0, stream>>>(h, B_ * D_);
  zero_f32<<<16,   256, 0, stream>>>((float*)h_bf, B_ * D_ / 2);

  // ---- embedding gather + hoisted input GEMM: gi = e @ W_ih^T + b_ih ----
  // MT=4: each wave computes 4 stacked C tiles sharing one B fragment.
  gather_embed<<<4096, 256, 0, stream>>>(tokens, emb, Ebf);
  gemm_bf16_wmma<4><<<dim3(12, B_ * L_ / 64), 256, 0, stream>>>(
      Ebf, WihBf, b_ih, gi_all, 3 * D_, D_, (3 * D_) / 16);

  // ---- sequential recurrence (M=16 -> MT=1) ----------------------------
  for (int t = 0; t < L_; ++t) {
    // gh = h @ W_hh^T + b_hh   (16 x 1536, K=512) : 96 tiles
    gemm_bf16_wmma<1><<<dim3(12, 1), 256, 0, stream>>>(
        h_bf, WhhBf, b_hh, gh, 3 * D_, D_, (3 * D_) / 16);
    // GRU gates -> h_new (f32 + bf16)
    gru_gates<<<(B_ * D_ + 255) / 256, 256, 0, stream>>>(gi_all, gh, h, h_bf, t);
    // [g|x|xp] = h_new @ [Wg;Wx;Wp]^T + [bg;bx;bp]  (16 x 768, K=512)
    gemm_bf16_wmma<1><<<dim3(6, 1), 256, 0, stream>>>(
        h_bf, WcatBf, bcat, gxp, 768, D_, 768 / 16);
    // memory read/write + layernorm -> xn (bf16)
    mem_update<<<B_, DX_, 0, stream>>>(gxp, Mbuf, ln_g, ln_b, xn_bf);
    // logits = xn @ Wo^T + bo  (16 x 10000, K=256), written straight to out
    gemm_bf16_wmma<1><<<dim3((V_ / 16 + 7) / 8, 1), 256, 0, stream>>>(
        xn_bf, WoBf, bo, out + (size_t)t * V_, (long)L_ * V_, DX_, V_ / 16);
  }
}
